// RNN_65472481460411
// MI455X (gfx1250) — matrix operand order repaired and checked by reading, not compiled, not run on hardware
//
#include <hip/hip_runtime.h>
#include <stdint.h>

// MI455X (gfx1250) 2-layer tanh RNN, latency-optimized.
// One wave32 per 16-batch tile; state kept transposed (hidden x batch) so the
// whole layer-2 recurrence (h2@W_hh2^T + h1@W_ih2^T + bias) is ONE
// v_wmma_f32_16x16x32_f16 per timestep (f32 accumulate, f16 only on the
// tanh-bounded state).
//
// Key trick: the constant A matrix's columns are PERMUTED so the B matrix can
// be rebuilt each step with zero cross-lane data movement:
//   K 0..7   = h2 rows 0..7   (lanes 0-15 hold these locally)
//   K 16..23 = h2 rows 8..15  (lanes 16-31 hold these locally)
//   K 24..27 = h1 rows 0..3   (mirrored in both halves)
//   K 8..15, 28..31 = ZERO COLUMNS in A -> B content there is irrelevant.
// So B build = 6 v_cvt_pk_rtz, no ds_swizzle on the recurrence path.
// Layer 1 (4x4) runs on VALU one step AHEAD to fill the WMMA hazard shadow.

typedef __attribute__((ext_vector_type(16))) _Float16 v16h;
typedef __attribute__((ext_vector_type(8)))  float    v8f;
typedef __attribute__((ext_vector_type(8)))  uint32_t u32x8;
typedef __attribute__((ext_vector_type(4)))  float    f32x4;
typedef __attribute__((ext_vector_type(2)))  float    f32x2;

#define T_LEN 4096

__device__ __forceinline__ float fast_tanh(float x) {
#if __has_builtin(__builtin_amdgcn_tanhf)
  return __builtin_amdgcn_tanhf(x);            // v_tanh_f32 (gfx1250 TRANS op)
#elif __has_builtin(__builtin_amdgcn_tanh_f32)
  return __builtin_amdgcn_tanh_f32(x);
#else
  float e = __builtin_amdgcn_exp2f(x * 2.8853900817779268f);
  return 1.0f - 2.0f * __builtin_amdgcn_rcpf(e + 1.0f);
#endif
}

// swap lane halves 0-15 <-> 16-31 (ds_swizzle group-of-32: xor=0x10,and=0x1f)
__device__ __forceinline__ float swz16f(float v) {
  int r = __builtin_amdgcn_ds_swizzle(__builtin_bit_cast(int, v), 0x401f);
  return __builtin_bit_cast(float, r);
}
__device__ __forceinline__ uint32_t pk16(float lo, float hi) {
  return __builtin_bit_cast(uint32_t, __builtin_amdgcn_cvt_pkrtz(lo, hi));
}

__global__ __launch_bounds__(32) void RNN_65472481460411_kernel(
    const float* __restrict__ x,
    const float* __restrict__ W_ih1, const float* __restrict__ W_hh1,
    const float* __restrict__ b_ih1, const float* __restrict__ b_hh1,
    const float* __restrict__ W_ih2, const float* __restrict__ W_hh2,
    const float* __restrict__ b_ih2, const float* __restrict__ b_hh2,
    const float* __restrict__ Wf,   const float* __restrict__ bf,
    float* __restrict__ out)
{
  const int lane = threadIdx.x;      // 0..31
  const int m    = lane & 15;        // row index / mirrored batch index
  const int hi   = (lane >> 4) & 1;  // lane half
  const int b    = blockIdx.x * 16 + m;

  // ---- layer-1 weights: uniform -> SGPRs ----
  float wih1[4], bias1[4], whh1[4][4];
#pragma unroll
  for (int v = 0; v < 4; ++v) {
    wih1[v]  = W_ih1[v];
    bias1[v] = b_ih1[v] + b_hh1[v];
#pragma unroll
    for (int k = 0; k < 4; ++k) whh1[v][k] = W_hh1[v * 4 + k];
  }

  // ---- constant A matrix: 16x32 f16, column-permuted (see header) ----
  // A as packed here (contiguous k order, not the ISA table order): lane m; K = 2v + 16*hi
  u32x8 a2u;
#pragma unroll
  for (int v = 0; v < 8; ++v) {
    int k0 = 2 * v + 16 * hi;
    float e[2];
#pragma unroll
    for (int j = 0; j < 2; ++j) {
      int k = k0 + j;
      float val = 0.0f;
      if (k < 8)               val = W_hh2[m * 16 + k];          // h2 rows 0-7
      else if (k >= 16 && k < 24) val = W_hh2[m * 16 + (k - 8)]; // h2 rows 8-15
      else if (k >= 24 && k < 28) val = W_ih2[m * 4 + (k - 24)]; // h1 rows 0-3
      e[j] = val;                                                // K 8-15,28-31 = 0
    }
    a2u[v] = pk16(e[0], e[1]);
  }
  const v16h a2 = __builtin_bit_cast(v16h, a2u);

  // ---- bias2 broadcast in C/D layout: vgpr v holds bias2[v + 8*hi] ----
  v8f cb;
#pragma unroll
  for (int v = 0; v < 8; ++v) cb[v] = b_ih2[v + 8 * hi] + b_hh2[v + 8 * hi];

  // ---- head coefficients in D layout ----
  float wf0[8], wf1[8];
#pragma unroll
  for (int v = 0; v < 8; ++v) {
    wf0[v] = Wf[v + 8 * hi];
    wf1[v] = Wf[16 + v + 8 * hi];
  }
  const float bf0 = bf[0], bf1 = bf[1];

  const float* xp = x + (size_t)b * T_LEN;
  float*       op = out + (size_t)b * (T_LEN * 2);

  f32x4 xq = *(const f32x4*)(xp);      // t = 0..3
  f32x4 xn = *(const f32x4*)(xp + 4);  // t = 4..7

  // ---- state: h1 is pipelined ONE STEP AHEAD (holds h1[t] entering step t) ----
  float h1s[4];
#pragma unroll
  for (int v = 0; v < 4; ++v)          // h1[0] = tanh(x0*W_ih1 + bias1)
    h1s[v] = fast_tanh(__builtin_fmaf(xq[0], wih1[v], bias1[v]));

  float t2[8] = {0.f, 0.f, 0.f, 0.f, 0.f, 0.f, 0.f, 0.f}; // h2 rows v(+8*hi)

  for (int t4 = 0; t4 < T_LEN; t4 += 4) {
    const f32x4 xc = xq;
    xq = xn;
    int nt = t4 + 8;
    if (nt > T_LEN - 4) nt = T_LEN - 4;
    xn = *(const f32x4*)(xp + nt);
    __builtin_prefetch(xp + ((t4 + 64 < T_LEN) ? t4 + 64 : t4), 0, 1);

#pragma unroll
    for (int s = 0; s < 4; ++s) {
      // ---- build B (32x16 f16) from local regs only: 6 cvt_pk, no shuffles ----
      u32x8 bu;
      bu[0] = pk16(t2[0], t2[1]);   // K 0-1  | K 16-17  (h2 rows, per half)
      bu[1] = pk16(t2[2], t2[3]);   // K 2-3  | K 18-19
      bu[2] = pk16(t2[4], t2[5]);   // K 4-5  | K 20-21
      bu[3] = pk16(t2[6], t2[7]);   // K 6-7  | K 22-23
      bu[4] = pk16(h1s[0], h1s[1]); // K 8-9  | K 24-25  (A cols 8-15 are zero)
      bu[5] = pk16(h1s[2], h1s[3]); // K 10-11| K 26-27
      bu[6] = 0u;                   // K 12-13| K 28-29  (zero A cols)
      bu[7] = 0u;                   // K 14-15| K 30-31
      const v16h bmat = __builtin_bit_cast(v16h, bu);

      // ---- layer 2: one WMMA = h2@W_hh2^T + h1@W_ih2^T + bias2 ----
      v8f d = __builtin_amdgcn_wmma_f32_16x16x32_f16(
          false, a2, false, bmat, (short)0, cb, false, false);

      // ---- layer 1 for step t+1 (independent of d; fills WMMA hazard shadow)
      {
        const float xnext = (s == 3) ? xq[0] : xc[s + 1]; // last iter: dead value
        float h1n[4];
#pragma unroll
        for (int v = 0; v < 4; ++v) {
          float a = __builtin_fmaf(xnext, wih1[v], bias1[v]);
          a = __builtin_fmaf(whh1[v][0], h1s[0], a);
          a = __builtin_fmaf(whh1[v][1], h1s[1], a);
          a = __builtin_fmaf(whh1[v][2], h1s[2], a);
          a = __builtin_fmaf(whh1[v][3], h1s[3], a);
          h1n[v] = fast_tanh(a);
        }
#pragma unroll
        for (int v = 0; v < 4; ++v) h1s[v] = h1n[v];
      }

      // ---- new h2 state ----
#pragma unroll
      for (int v = 0; v < 8; ++v) t2[v] = fast_tanh(d[v]);

      // ---- head: out[b][t][c] = Wf[c,:] . h2[:,b] + bf[c] (off critical path)
      float acc0 = 0.f, acc1 = 0.f;
#pragma unroll
      for (int v = 0; v < 8; ++v) {
        acc0 = __builtin_fmaf(wf0[v], t2[v], acc0);
        acc1 = __builtin_fmaf(wf1[v], t2[v], acc1);
      }
      const float r0 = acc0 + swz16f(acc0) + bf0; // combine both lane-halves
      const float r1 = acc1 + swz16f(acc1) + bf1;
      if (!hi) {
        f32x2 o; o.x = r0; o.y = r1;
        *(f32x2*)(op + (size_t)(t4 + s) * 2) = o;
      }
    }
  }
}

extern "C" void kernel_launch(void* const* d_in, const int* in_sizes, int n_in,
                              void* d_out, int out_size, void* d_ws, size_t ws_size,
                              hipStream_t stream) {
  const float* x     = (const float*)d_in[0];
  const float* W_ih1 = (const float*)d_in[1];
  const float* W_hh1 = (const float*)d_in[2];
  const float* b_ih1 = (const float*)d_in[3];
  const float* b_hh1 = (const float*)d_in[4];
  const float* W_ih2 = (const float*)d_in[5];
  const float* W_hh2 = (const float*)d_in[6];
  const float* b_ih2 = (const float*)d_in[7];
  const float* b_hh2 = (const float*)d_in[8];
  const float* Wf    = (const float*)d_in[9];
  const float* bf    = (const float*)d_in[10];

  const int B = in_sizes[0] / T_LEN;   // x is (B,1,T) -> 512
  const int nblocks = B / 16;          // one wave32 per 16-batch tile

  RNN_65472481460411_kernel<<<dim3(nblocks), dim3(32), 0, stream>>>(
      x, W_ih1, W_hh1, b_ih1, b_hh1, W_ih2, W_hh2, b_ih2, b_hh2, Wf, bf,
      (float*)d_out);
}
